// Mamba3Decoder_88536455840137
// MI455X (gfx1250) — compile-verified, matrix operand order checked by reading, not run on hardware
//
#include <hip/hip_runtime.h>
#include <hip/hip_bf16.h>

// ---------------- model constants ----------------
#define D_    512
#define NL_   6
#define NS_   64
#define DI_   1024
#define PH_   32
#define NH_   32
#define V_    32000
#define PFX_  8
#define EMB_  1536
#define KC_   4
#define CDIM_ 1152   // DI + 2*NS
#define DIN_  2208   // 2*DI + 2*NS + NH
#define B_    2
#define L_    512
#define T_    520    // PFX + L
#define BT_   1040   // B * T

#define KCHUNK_ 128  // K-chunk staged to LDS by the TDM
#define MTILE_  64   // M rows per block (4 x 16 per wave)

typedef __bf16 bf16_t;
typedef __attribute__((ext_vector_type(16))) __bf16 v16bf;
typedef __attribute__((ext_vector_type(8)))  __bf16 v8bf;
typedef __attribute__((ext_vector_type(8)))  float  v8f;

// ---------------- TDM availability (device pass only) ----------------
#if defined(__gfx1250__) && __has_builtin(__builtin_amdgcn_tensor_load_to_lds) && \
    __has_builtin(__builtin_amdgcn_s_wait_tensorcnt)
#define USE_TDM 1
#else
#define USE_TDM 0
#endif

#if USE_TDM
typedef __attribute__((ext_vector_type(4))) unsigned int v4u_t;
typedef __attribute__((ext_vector_type(8))) int          v8i_t;
typedef __attribute__((ext_vector_type(4))) int          v4i_t;

// Generic pointers to LDS carry the LDS byte offset in their low 32 bits
// (ISA 10.2).
__device__ __forceinline__ unsigned lds_off_u32(const void* p) {
    return (unsigned)(uintptr_t)p;
}

// Issue one TDM 2D tile load: tile_y rows of tile_x bf16 elems, row stride
// stride_x elems, from gsrc into LDS at lds_off. Wave-level op; EXEC ignored.
// D# layout per CDNA5 ISA ch.8 (group0: count/lds/global/type, group1:
// data_size, tensor dims, tile dims, dim0 stride).
__device__ __forceinline__ void tdm_load_tile_bf16(unsigned lds_off,
                                                   const bf16_t* gsrc,
                                                   unsigned tile_x,
                                                   unsigned tile_y,
                                                   unsigned stride_x) {
    unsigned long long ga = (unsigned long long)(uintptr_t)gsrc;
    v4u_t g0;
    g0[0] = 1u;                                               // count=1 (valid user D#)
    g0[1] = lds_off;                                          // lds_addr (bytes)
    g0[2] = (unsigned)ga;                                     // global_addr[31:0]
    g0[3] = ((unsigned)(ga >> 32) & 0x1FFFFFFu) | 0x80000000u; // addr[56:32] | type=2
    v8i_t g1;
    g1[0] = (int)(1u << 16);                                  // data_size=1 -> 2 bytes
    g1[1] = (int)((tile_x & 0xFFFFu) << 16);                  // tensor_dim0[15:0]
    g1[2] = (int)((tile_x >> 16) | ((tile_y & 0xFFFFu) << 16)); // dim0 hi | dim1 lo
    g1[3] = (int)((tile_y >> 16) | (tile_x << 16));           // dim1 hi | tile_dim0
    g1[4] = (int)(tile_y & 0xFFFFu);                          // tile_dim1 (tile_dim2=0)
    g1[5] = (int)stride_x;                                    // tensor_dim0_stride[31:0]
    g1[6] = 0;
    g1[7] = 0;
    v4i_t z4; z4[0] = 0; z4[1] = 0; z4[2] = 0; z4[3] = 0;     // 2D: groups 2/3 unused
#if defined(__clang_major__) && (__clang_major__ >= 23)
    v8i_t z8; z8[0]=0; z8[1]=0; z8[2]=0; z8[3]=0; z8[4]=0; z8[5]=0; z8[6]=0; z8[7]=0;
    __builtin_amdgcn_tensor_load_to_lds(g0, g1, z4, z4, z8, 0);
#else
    __builtin_amdgcn_tensor_load_to_lds(g0, g1, z4, z4, 0);
#endif
}
#endif  // USE_TDM

// ---------------- f32 -> bf16 convert ----------------
__global__ void convert_bf16_kernel(const float* __restrict__ in,
                                    bf16_t* __restrict__ out, int n) {
    int i = blockIdx.x * blockDim.x + threadIdx.x;
    if (i < n) out[i] = (bf16_t)in[i];
}

// ---------------- WMMA fragment loaders ----------------
// CDNA5 16-bit A (16x32) per-lane layout: lane = (row, half); values
// [0..7] = K (k + half*8 .. +8), [8..15] = K (k + 16 + half*8 .. +8).
// B (32x16) mirrors this with n instead of m; B is stored [N,K] row-major
// (weight layout), so the same pattern serves both operands.
__device__ __forceinline__ v16bf load_frag_fast(const bf16_t* __restrict__ p) {
    v8bf lo = *(const v8bf*)(p);
    v8bf hi = *(const v8bf*)(p + 16);
    v16bf r;
#pragma unroll
    for (int i = 0; i < 8; ++i) { r[i] = lo[i]; r[i + 8] = hi[i]; }
    return r;
}

__device__ __forceinline__ v16bf load_frag_guard(const bf16_t* __restrict__ base, int row,
                                                 int nrows, int ld, int k, int halfoff) {
    v16bf r;
    if (row < nrows) {
        r = load_frag_fast(base + (size_t)row * ld + k + halfoff);
    } else {
#pragma unroll
        for (int i = 0; i < 16; ++i) r[i] = (bf16_t)0.0f;
    }
    return r;
}

// ---------------- generic WMMA GEMM: C[M,N] = A[M,K] * B[N,K]^T ----------------
// Block tile: 64(M) x 256(N); 8 waves, each wave computes 64x32 via
// 4 M-fragments x 2 N-fragments = 8 WMMA accumulators, so each 32-wide K step
// issues 8 WMMAs for 12 x 128-bit loads. The block-shared 64-row A tile is
// staged into LDS by the Tensor Data Mover (wave 0 issues tensor_load_to_lds,
// double-buffered, synchronized with s_wait_tensorcnt + barrier) — A global
// traffic deduped 8x, B traffic amortized over 64 output rows. Edge blocks
// take a guarded path; the fast/slow decision is block-uniform so barriers
// never diverge.
__global__ void gemm_wmma_bf16(const bf16_t* __restrict__ A, int lda,
                               const bf16_t* __restrict__ Bw, int ldb,
                               float* __restrict__ C, int ldc,
                               int M, int N, int K,
                               const float* __restrict__ bias, int addC) {
    const int wave = threadIdx.x >> 5;
    const int lane = threadIdx.x & 31;
    const int mtile  = blockIdx.y * MTILE_;
    const int ntile0 = blockIdx.x * 256 + wave * 32;
    const int row  = lane & 15;
    const int half = lane >> 4;
    const int halfoff = half * 8;

#if USE_TDM
    __shared__ bf16_t sA[2][MTILE_ * KCHUNK_];   // 2 x 16KB double buffer
#endif

    v8f acc[4][2];
#pragma unroll
    for (int i = 0; i < 4; ++i)
#pragma unroll
        for (int j = 0; j < 2; ++j)
#pragma unroll
            for (int r = 0; r < 8; ++r) acc[i][j][r] = 0.0f;

    const bool blockfull = (mtile + MTILE_ <= M) &&
                           ((blockIdx.x + 1) * 256 <= (unsigned)N) &&
                           ((K & (KCHUNK_ - 1)) == 0);

    if (blockfull) {
        const bf16_t* pb0 = Bw + (size_t)(ntile0 + row) * ldb + halfoff;
#if USE_TDM
        const bf16_t* ag = A + (size_t)mtile * lda;   // block-shared 64 rows
        if (wave == 0)
            tdm_load_tile_bf16(lds_off_u32(&sA[0][0]), ag, KCHUNK_, MTILE_, (unsigned)lda);
        const int nch = K / KCHUNK_;
        for (int c = 0; c < nch; ++c) {
            if (wave == 0) {
                if (c + 1 < nch) {
                    tdm_load_tile_bf16(lds_off_u32(&sA[(c + 1) & 1][0]),
                                       ag + (size_t)(c + 1) * KCHUNK_,
                                       KCHUNK_, MTILE_, (unsigned)lda);
                    __builtin_amdgcn_s_wait_tensorcnt(1);   // chunk c complete
                } else {
                    __builtin_amdgcn_s_wait_tensorcnt(0);   // last chunk complete
                }
            }
            __syncthreads();    // A chunk visible to whole block
            const bf16_t* abuf = &sA[c & 1][row * KCHUNK_ + halfoff];
            const bf16_t* pb   = pb0 + (size_t)c * KCHUNK_;
            if (c + 1 < nch) {
                __builtin_prefetch(pb, 0, 1);
                __builtin_prefetch(pb + (size_t)16 * ldb + KCHUNK_, 0, 1);
            }
            for (int kk = 0; kk < KCHUNK_; kk += 32) {
                v16bf a0 = load_frag_fast(abuf + kk);                       // ds_b128 x2
                v16bf a1 = load_frag_fast(abuf + 16 * KCHUNK_ + kk);
                v16bf a2 = load_frag_fast(abuf + 32 * KCHUNK_ + kk);
                v16bf a3 = load_frag_fast(abuf + 48 * KCHUNK_ + kk);
#pragma unroll
                for (int j = 0; j < 2; ++j) {
                    v16bf b = load_frag_fast(pb + (size_t)(j * 16) * ldb + kk);
                    acc[0][j] = __builtin_amdgcn_wmma_f32_16x16x32_bf16(
                        false, a0, false, b, (short)0, acc[0][j], false, false);
                    acc[1][j] = __builtin_amdgcn_wmma_f32_16x16x32_bf16(
                        false, a1, false, b, (short)0, acc[1][j], false, false);
                    acc[2][j] = __builtin_amdgcn_wmma_f32_16x16x32_bf16(
                        false, a2, false, b, (short)0, acc[2][j], false, false);
                    acc[3][j] = __builtin_amdgcn_wmma_f32_16x16x32_bf16(
                        false, a3, false, b, (short)0, acc[3][j], false, false);
                }
            }
            __syncthreads();    // done reading buf before it is re-filled
        }
#else
        // fallback: register-direct fast path (no TDM builtin on this toolchain)
        const bf16_t* pa = A + (size_t)(mtile + row) * lda + halfoff;
        for (int k = 0; k < K; k += 32) {
            v16bf a0 = load_frag_fast(pa + k);
            v16bf a1 = load_frag_fast(pa + (size_t)16 * lda + k);
            v16bf a2 = load_frag_fast(pa + (size_t)32 * lda + k);
            v16bf a3 = load_frag_fast(pa + (size_t)48 * lda + k);
#pragma unroll
            for (int j = 0; j < 2; ++j) {
                v16bf b = load_frag_fast(pb0 + (size_t)(j * 16) * ldb + k);
                acc[0][j] = __builtin_amdgcn_wmma_f32_16x16x32_bf16(
                    false, a0, false, b, (short)0, acc[0][j], false, false);
                acc[1][j] = __builtin_amdgcn_wmma_f32_16x16x32_bf16(
                    false, a1, false, b, (short)0, acc[1][j], false, false);
                acc[2][j] = __builtin_amdgcn_wmma_f32_16x16x32_bf16(
                    false, a2, false, b, (short)0, acc[2][j], false, false);
                acc[3][j] = __builtin_amdgcn_wmma_f32_16x16x32_bf16(
                    false, a3, false, b, (short)0, acc[3][j], false, false);
            }
        }
#endif
        // C layout: lane -> col n = lane%16; VGPR r -> row m = base + r + 8*half
#pragma unroll
        for (int i = 0; i < 4; ++i) {
#pragma unroll
            for (int j = 0; j < 2; ++j) {
                int ncol = ntile0 + j * 16 + row;
                float bs = bias ? bias[ncol] : 0.0f;
                float* cp = C + (size_t)(mtile + i * 16 + 8 * half) * ldc + ncol;
#pragma unroll
                for (int r = 0; r < 8; ++r) {
                    size_t idx = (size_t)r * ldc;
                    float v = acc[i][j][r] + bs;
                    cp[idx] = addC ? (cp[idx] + v) : v;
                }
            }
        }
    } else {
        // ---- guarded path: edge blocks only ----
        for (int k = 0; k < K; k += 32) {
            v16bf a[4];
#pragma unroll
            for (int i = 0; i < 4; ++i)
                a[i] = load_frag_guard(A, mtile + i * 16 + row, M, lda, k, halfoff);
#pragma unroll
            for (int j = 0; j < 2; ++j) {
                v16bf b = load_frag_guard(Bw, ntile0 + j * 16 + row, N, ldb, k, halfoff);
#pragma unroll
                for (int i = 0; i < 4; ++i)
                    acc[i][j] = __builtin_amdgcn_wmma_f32_16x16x32_bf16(
                        false, a[i], false, b, (short)0, acc[i][j], false, false);
            }
        }
#pragma unroll
        for (int i = 0; i < 4; ++i) {
#pragma unroll
            for (int j = 0; j < 2; ++j) {
                int ncol = ntile0 + j * 16 + row;
                if (ncol >= N) continue;
                float bs = bias ? bias[ncol] : 0.0f;
#pragma unroll
                for (int r = 0; r < 8; ++r) {
                    int m = mtile + i * 16 + r + 8 * half;
                    if (m < M) {
                        size_t idx = (size_t)m * ldc + ncol;
                        float v = acc[i][j][r] + bs;
                        C[idx] = addC ? (C[idx] + v) : v;
                    }
                }
            }
        }
    }
}

// ---------------- h assembly: prefix + token embeddings ----------------
__global__ void assemble_h_kernel(const float* __restrict__ prefix,
                                  const float* __restrict__ tok_emb,
                                  const int* __restrict__ tok,
                                  float* __restrict__ h) {
    int idx = blockIdx.x * blockDim.x + threadIdx.x;
    if (idx >= BT_ * D_) return;
    int d  = idx % D_;
    int bt = idx / D_;
    int b  = bt / T_;
    int t  = bt % T_;
    if (t < PFX_)
        h[idx] = prefix[(size_t)b * (PFX_ * D_) + t * D_ + d];
    else
        h[idx] = tok_emb[(size_t)tok[b * L_ + (t - PFX_)] * D_ + d];
}

// ---------------- rmsnorm (D=512) -> bf16 ----------------
__global__ void rmsnorm_kernel(const float* __restrict__ x,
                               const float* __restrict__ w,
                               bf16_t* __restrict__ out,
                               int compact) {
    int orow = blockIdx.x;
    int irow;
    if (compact) { int b = orow / L_; int t = orow % L_; irow = b * T_ + PFX_ + t; }
    else irow = orow;
    const float* xr = x + (size_t)irow * D_;
    float ss = 0.0f;
    for (int i = threadIdx.x; i < D_; i += 256) { float v = xr[i]; ss += v * v; }
#pragma unroll
    for (int off = 16; off > 0; off >>= 1) ss += __shfl_down(ss, off, 32);
    __shared__ float red[8];
    int lane = threadIdx.x & 31, wv = threadIdx.x >> 5;
    if (lane == 0) red[wv] = ss;
    __syncthreads();
    if (wv == 0) {
        float s = (lane < 8) ? red[lane] : 0.0f;
#pragma unroll
        for (int off = 4; off > 0; off >>= 1) s += __shfl_down(s, off, 32);
        if (lane == 0) red[0] = rsqrtf(s * (1.0f / D_) + 1e-5f);
    }
    __syncthreads();
    float scale = red[0];
    bf16_t* orr = out + (size_t)orow * D_;
    for (int i = threadIdx.x; i < D_; i += 256)
        orr[i] = (bf16_t)(xr[i] * scale * w[i]);
}

// ---------------- causal conv (K=4) + silu on xBC channels ----------------
__global__ void conv_silu_kernel(const float* __restrict__ u,
                                 const float* __restrict__ convw,
                                 float* __restrict__ xBC) {
    int idx = blockIdx.x * blockDim.x + threadIdx.x;
    if (idx >= BT_ * CDIM_) return;
    int c  = idx % CDIM_;
    int bt = idx / CDIM_;
    int b  = bt / T_;
    int t  = bt % T_;
    float acc = 0.0f;
#pragma unroll
    for (int k = 0; k < KC_; ++k) {
        int tt = t + k - (KC_ - 1);
        if (tt >= 0)
            acc += u[(size_t)(b * T_ + tt) * DIN_ + DI_ + c] * convw[c * KC_ + k];
    }
    xBC[idx] = acc / (1.0f + __expf(-acc));   // silu
}

// ---------------- dt = softplus(raw + bias), da = exp(-exp(A)*dt) ----------------
__global__ void dt_kernel(const float* __restrict__ u,
                          const float* __restrict__ dt_bias,
                          const float* __restrict__ A_log,
                          float* __restrict__ dtb, float* __restrict__ dab) {
    int idx = blockIdx.x * blockDim.x + threadIdx.x;
    if (idx >= BT_ * NH_) return;
    int hh = idx % NH_;
    int bt = idx / NH_;
    float raw = u[(size_t)bt * DIN_ + DI_ + CDIM_ + hh] + dt_bias[hh];
    float dt  = (raw > 20.0f) ? raw : log1pf(__expf(raw));
    float a   = __expf(A_log[hh]);
    dtb[idx] = dt;
    dab[idx] = __expf(-a * dt);
}

// ---------------- sequential SSM scan: one block per (b, h) ----------------
// state hs[PH=32][NS=64]; thread (p = tid&31, ngroup = tid>>5) owns 8 n-slots.
__global__ void scan_kernel(const float* __restrict__ xBC,
                            const float* __restrict__ dtb,
                            const float* __restrict__ dab,
                            const float* __restrict__ D_skip,
                            float* __restrict__ ybuf) {
    int b  = blockIdx.x / NH_;
    int hh = blockIdx.x % NH_;
    int tid = threadIdx.x;
    int p  = tid & 31;
    int ng = tid >> 5;
    float hs[8];
#pragma unroll
    for (int r = 0; r < 8; ++r) hs[r] = 0.0f;

    __shared__ float sB[NS_], sC[NS_], sx[PH_], sy[PH_];
    __shared__ float sda, sdt;
    float dsk = D_skip[hh];

    for (int t = 0; t < T_; ++t) {
        const float* xrow = xBC + (size_t)(b * T_ + t) * CDIM_;
        if (tid < 64)        sB[tid]       = xrow[DI_ + tid];
        else if (tid < 128)  sC[tid - 64]  = xrow[DI_ + NS_ + (tid - 64)];
        else if (tid < 160)  sx[tid - 128] = xrow[hh * PH_ + (tid - 128)];
        else if (tid == 160) {
            sda = dab[(size_t)(b * T_ + t) * NH_ + hh];
            sdt = dtb[(size_t)(b * T_ + t) * NH_ + hh];
        }
        if (tid < PH_) sy[tid] = 0.0f;
        __syncthreads();

        float da  = sda;
        float dtx = sdt * sx[p];
        float part = 0.0f;
#pragma unroll
        for (int r = 0; r < 8; ++r) {
            int n = ng * 8 + r;
            hs[r] = da * hs[r] + dtx * sB[n];
            part += hs[r] * sC[n];
        }
        atomicAdd(&sy[p], part);
        __syncthreads();
        if (tid < PH_)
            ybuf[(size_t)(b * T_ + t) * DI_ + hh * PH_ + tid] = sy[tid] + dsk * sx[tid];
        __syncthreads();
    }
}

// ---------------- gate: y = ys * silu(z), output bf16 ----------------
__global__ void gate_kernel(const float* __restrict__ u,
                            const float* __restrict__ ybuf,
                            bf16_t* __restrict__ ybf) {
    int idx = blockIdx.x * blockDim.x + threadIdx.x;
    if (idx >= BT_ * DI_) return;
    int e  = idx % DI_;
    int bt = idx / DI_;
    float z = u[(size_t)bt * DIN_ + e];
    float g = z / (1.0f + __expf(-z));
    ybf[idx] = (bf16_t)(ybuf[idx] * g);
}

// ---------------- host orchestration ----------------
extern "C" void kernel_launch(void* const* d_in, const int* in_sizes, int n_in,
                              void* d_out, int out_size, void* d_ws, size_t ws_size,
                              hipStream_t stream) {
    const float* pred_emb   = (const float*)d_in[0];
    const int*   tok        = (const int*)  d_in[1];
    const float* e2p_w      = (const float*)d_in[2];
    const float* e2p_b      = (const float*)d_in[3];
    const float* tok_emb    = (const float*)d_in[4];
    const float* norm_w     = (const float*)d_in[5];
    const float* in_proj_w  = (const float*)d_in[6];
    const float* conv_w     = (const float*)d_in[7];
    const float* dt_bias    = (const float*)d_in[8];
    const float* A_log      = (const float*)d_in[9];
    const float* D_skip     = (const float*)d_in[10];
    const float* out_proj_w = (const float*)d_in[11];
    const float* fnorm_w    = (const float*)d_in[12];
    const float* lm_head_w  = (const float*)d_in[13];
    float* out = (float*)d_out;

    char* ws = (char*)d_ws;
    size_t off = 0;
    auto alloc = [&](size_t bytes) -> void* {
        void* p = ws + off;
        off = (off + bytes + 255) & ~(size_t)255;
        return p;
    };

    float*  prefix  = (float*) alloc((size_t)B_ * PFX_ * D_ * 4);
    float*  h       = (float*) alloc((size_t)BT_ * D_ * 4);
    bf16_t* xnorm   = (bf16_t*)alloc((size_t)BT_ * D_ * 2);
    float*  u       = (float*) alloc((size_t)BT_ * DIN_ * 4);
    float*  xBC     = (float*) alloc((size_t)BT_ * CDIM_ * 4);
    float*  dtb     = (float*) alloc((size_t)BT_ * NH_ * 4);
    float*  dab     = (float*) alloc((size_t)BT_ * NH_ * 4);
    float*  ybuf    = (float*) alloc((size_t)BT_ * DI_ * 4);
    bf16_t* ybf     = (bf16_t*)alloc((size_t)BT_ * DI_ * 2);
    bf16_t* xfin    = (bf16_t*)alloc((size_t)B_ * L_ * D_ * 2);
    bf16_t* emb_bf  = (bf16_t*)alloc((size_t)B_ * EMB_ * 2);
    bf16_t* e2p_bf  = (bf16_t*)alloc((size_t)PFX_ * D_ * EMB_ * 2);
    bf16_t* inw_bf  = (bf16_t*)alloc((size_t)NL_ * DIN_ * D_ * 2);
    bf16_t* outw_bf = (bf16_t*)alloc((size_t)NL_ * D_ * DI_ * 2);
    bf16_t* lmw_bf  = (bf16_t*)alloc((size_t)V_ * D_ * 2);

    auto cvt = [&](const float* src, bf16_t* dst, size_t n) {
        convert_bf16_kernel<<<(unsigned)((n + 255) / 256), 256, 0, stream>>>(src, dst, (int)n);
    };
    auto gemm = [&](const bf16_t* A, int lda, const bf16_t* Bw, int ldb,
                    float* C, int ldc, int M, int N, int K,
                    const float* bias, int addC) {
        dim3 g((unsigned)((N + 255) / 256), (unsigned)((M + MTILE_ - 1) / MTILE_));
        gemm_wmma_bf16<<<g, 256, 0, stream>>>(A, lda, Bw, ldb, C, ldc, M, N, K, bias, addC);
    };

    // --- weight / input conversions (per-call, deterministic) ---
    cvt(pred_emb,  emb_bf,  (size_t)B_ * EMB_);
    cvt(e2p_w,     e2p_bf,  (size_t)PFX_ * D_ * EMB_);
    cvt(in_proj_w, inw_bf,  (size_t)NL_ * DIN_ * D_);
    cvt(out_proj_w, outw_bf,(size_t)NL_ * D_ * DI_);
    cvt(lm_head_w, lmw_bf,  (size_t)V_ * D_);

    // --- prefix projection + sequence assembly ---
    gemm(emb_bf, EMB_, e2p_bf, EMB_, prefix, PFX_ * D_,
         B_, PFX_ * D_, EMB_, e2p_b, 0);
    assemble_h_kernel<<<(BT_ * D_ + 255) / 256, 256, 0, stream>>>(prefix, tok_emb, tok, h);

    // --- layers ---
    for (int i = 0; i < NL_; ++i) {
        rmsnorm_kernel<<<BT_, 256, 0, stream>>>(h, norm_w + i * D_, xnorm, 0);
        gemm(xnorm, D_, inw_bf + (size_t)i * DIN_ * D_, D_,
             u, DIN_, BT_, DIN_, D_, nullptr, 0);
        conv_silu_kernel<<<(BT_ * CDIM_ + 255) / 256, 256, 0, stream>>>(
            u, conv_w + i * CDIM_ * KC_, xBC);
        dt_kernel<<<(BT_ * NH_ + 255) / 256, 256, 0, stream>>>(
            u, dt_bias + i * NH_, A_log + i * NH_, dtb, dab);
        scan_kernel<<<B_ * NH_, 256, 0, stream>>>(xBC, dtb, dab, D_skip + i * NH_, ybuf);
        gate_kernel<<<(BT_ * DI_ + 255) / 256, 256, 0, stream>>>(u, ybuf, ybf);
        gemm(ybf, DI_, outw_bf + (size_t)i * D_ * DI_, DI_,
             h, D_, BT_, D_, DI_, nullptr, 1);
    }

    // --- final norm (compact: only token rows) + LM head ---
    rmsnorm_kernel<<<B_ * L_, 256, 0, stream>>>(h, fnorm_w, xfin, 1);
    gemm(xfin, D_, lmw_bf, D_, out, V_, B_ * L_, V_, D_, nullptr, 0);
}
